// CrossAndSelfAttention_57062935495373
// MI455X (gfx1250) — matrix operand order repaired and checked by reading, not compiled, not run on hardware
//
#include <hip/hip_runtime.h>

typedef __bf16 bf16;
typedef __attribute__((ext_vector_type(16))) __bf16 v16bf;
typedef __attribute__((ext_vector_type(8)))  __bf16 v8bf;
typedef __attribute__((ext_vector_type(8)))  float  v8f;
typedef __attribute__((ext_vector_type(4)))  int    v4i;

#define NTOK   65536
#define DMODEL 640
#define DHEAD  64
#define NHEADS 10

// CDNA5 async global->LDS copies (ASYNCcnt), guarded so compile never breaks.
#if defined(__has_builtin)
#if __has_builtin(__builtin_amdgcn_global_load_async_to_lds_b128) && \
    __has_builtin(__builtin_amdgcn_s_wait_asynccnt)
#define HAVE_ASYNC_LDS 1
#endif
#endif

#ifdef HAVE_ASYNC_LDS
typedef __attribute__((address_space(1))) v4i* as1_v4i;
typedef __attribute__((address_space(3))) v4i* as3_v4i;
__device__ __forceinline__ void async_cp16(const bf16* g, bf16* l) {
  __builtin_amdgcn_global_load_async_to_lds_b128(
      (as1_v4i)(void*)g, (as3_v4i)(void*)l, 0, 0);
}
#endif

// ---------------------------------------------------------------------------
// WMMA helpers (CDNA5: D = A(16x32 bf16) * B(32x16 bf16) + C(16x16 f32))
// ---------------------------------------------------------------------------
__device__ __forceinline__ v8f wmma_bf16(v16bf a, v16bf b, v8f c) {
  return __builtin_amdgcn_wmma_f32_16x16x32_bf16(false, a, false, b,
                                                 (short)0, c, false, false);
}

__device__ __forceinline__ v16bf cat8(v8bf lo, v8bf hi) {
  v16bf r;
#pragma unroll
  for (int i = 0; i < 8; ++i) { r[i] = lo[i]; r[i + 8] = hi[i]; }
  return r;
}

// A-tile 16x32 from row-major src (row stride ld elements), rows = lane&15.
// ISA layout: lanes 0-15 hold K {0..7,16..23}, lanes 16-31 hold K {8..15,24..31}.
__device__ __forceinline__ v16bf load_a_tile(const bf16* src, int ld, int lane) {
  const bf16* p = src + (size_t)(lane & 15) * ld + ((lane >> 4) << 3);
  return cat8(*(const v8bf*)p, *(const v8bf*)(p + 16));
}

// B-tile 32x16 where B[k][n] = src[n][k] (src row-major, row = output col n).
// ISA layout: lanes 0-15 hold K {0..7,16..23}, lanes 16-31 hold K {8..15,24..31}.
__device__ __forceinline__ v16bf load_b_tile(const bf16* src, int ld, int lane) {
  const bf16* p = src + (size_t)(lane & 15) * ld + ((lane >> 4) << 3);
  return cat8(*(const v8bf*)p, *(const v8bf*)(p + 16));
}

// ---------------------------------------------------------------------------
// Generic GEMM: C[M, nout] = A[M, KD](bf16) @ W[nout, KD]^T(bf16) + bias
// grid = (M/256, nout/64), block = 256 (8 waves). Each wave owns a 32x64
// C tile: two 16-row A tiles share the 4 B tiles (1.5 b128 loads per WMMA).
// KD is a template constant so the k-loop fully unrolls: no rotation copies,
// and the scheduler hoists next-step loads across the WMMA stream.
// ---------------------------------------------------------------------------
template <int KD>
__global__ __launch_bounds__(256) void gemm_bias_kernel(
    const bf16* __restrict__ A, const bf16* __restrict__ W,
    const float* __restrict__ bias, bf16* __restrict__ Cb,
    float* __restrict__ Cf, int ldc, int relu) {
  const int lane = threadIdx.x & 31;
  const int wave = threadIdx.x >> 5;
  const int r0 = blockIdx.x * 256 + wave * 32;
  const int c0 = blockIdx.y * 64;

  v8f acc[2][4];
#pragma unroll
  for (int i = 0; i < 2; ++i)
#pragma unroll
    for (int t = 0; t < 4; ++t)
#pragma unroll
      for (int v = 0; v < 8; ++v) acc[i][t][v] = 0.f;

  const bf16* arow0 = A + (size_t)r0 * KD;
  const bf16* arow1 = A + (size_t)(r0 + 16) * KD;
  const bf16* wrow = W + (size_t)c0 * KD;

#pragma unroll
  for (int k = 0; k < KD; k += 32) {
    v16bf a0 = load_a_tile(arow0 + k, KD, lane);
    v16bf a1 = load_a_tile(arow1 + k, KD, lane);
#pragma unroll
    for (int t = 0; t < 4; ++t) {
      v16bf b = load_b_tile(wrow + (size_t)(t * 16) * KD + k, KD, lane);
      acc[0][t] = wmma_bf16(a0, b, acc[0][t]);
      acc[1][t] = wmma_bf16(a1, b, acc[1][t]);
    }
  }

  const int lh = lane >> 4;
  const int cl = lane & 15;
#pragma unroll
  for (int t = 0; t < 4; ++t) {
    const int col = c0 + t * 16 + cl;
    const float bv = bias[col];
#pragma unroll
    for (int i = 0; i < 2; ++i)
#pragma unroll
      for (int v = 0; v < 8; ++v) {
        float x = acc[i][t][v] + bv;
        if (relu) x = fmaxf(x, 0.f);
        const size_t row = (size_t)(r0 + i * 16 + v + lh * 8);
        if (Cb) Cb[row * (size_t)ldc + col] = (bf16)x;
        if (Cf) Cf[row * (size_t)ldc + col] = x;
      }
  }
}

// ---------------------------------------------------------------------------
// Flash attention: one head per blockIdx.y, 64 query rows per block (4 waves).
// Double-buffered 64-key K/V tiles in LDS; K staged via async global->LDS
// copies (ASYNCcnt), V transposed through registers with the LDS stores
// deferred until after the WMMA work so the load latency hides under compute.
// ---------------------------------------------------------------------------
__global__ __launch_bounds__(128) void flash_attn_kernel(
    const bf16* __restrict__ Q, const bf16* __restrict__ K,
    const bf16* __restrict__ V, bf16* __restrict__ O, float scale) {
  __shared__ bf16 Kt[2][64 * 64];      // [key][d]   row-major
  __shared__ bf16 Vt[2][64 * 64];      // [d][key]   transposed
  __shared__ bf16 Pt[4][16 * 64];      // per-wave P scratch [row][key]

  const int tid = threadIdx.x;
  const int lane = tid & 31;
  const int wave = tid >> 5;
  const int h = blockIdx.y;
  const int q0 = blockIdx.x * 64;
  const int col0 = h * DHEAD;

  // Q A-tiles for this wave (rows q0+wave*16 .. +15), kept resident
  const bf16* qbase = Q + (size_t)(q0 + wave * 16) * DMODEL + col0;
  v16bf qa0 = load_a_tile(qbase, DMODEL, lane);        // d = 0..31
  v16bf qa1 = load_a_tile(qbase + 32, DMODEL, lane);   // d = 32..63

  v8f o[4];
  float m[8], l[8];
#pragma unroll
  for (int v = 0; v < 8; ++v) { m[v] = -1e30f; l[v] = 0.f; }
#pragma unroll
  for (int t = 0; t < 4; ++t)
#pragma unroll
    for (int v = 0; v < 8; ++v) o[t][v] = 0.f;

  // prologue: stage tile 0 into buffer 0
#pragma unroll
  for (int r = 0; r < 4; ++r) {
    const int idx = tid + r * 128;       // 512 chunks of 8 bf16
    const int row = idx >> 3;
    const int c8 = (idx & 7) << 3;
    const size_t g = (size_t)row * DMODEL + col0 + c8;
#ifdef HAVE_ASYNC_LDS
    async_cp16(&K[g], &Kt[0][row * 64 + c8]);
#else
    *(v8bf*)&Kt[0][row * 64 + c8] = *(const v8bf*)&K[g];
#endif
    v8bf vv = *(const v8bf*)&V[g];
#pragma unroll
    for (int i = 0; i < 8; ++i) Vt[0][(c8 + i) * 64 + row] = vv[i];
  }

  for (int kbase = 0, buf = 0; kbase < NTOK; kbase += 64, buf ^= 1) {
#ifdef HAVE_ASYNC_LDS
    __builtin_amdgcn_s_wait_asynccnt(0);
#endif
    __syncthreads();   // staging of `buf` visible; all waves done with buf^1

    // issue staging for the next tile into buf^1 (K async; V into registers)
    const bool has_next = (kbase + 64) < NTOK;
    v8bf vreg[4];
    if (has_next) {
#pragma unroll
      for (int r = 0; r < 4; ++r) {
        const int idx = tid + r * 128;
        const int row = idx >> 3;
        const int c8 = (idx & 7) << 3;
        const size_t g = (size_t)(kbase + 64 + row) * DMODEL + col0 + c8;
#ifdef HAVE_ASYNC_LDS
        async_cp16(&K[g], &Kt[buf ^ 1][row * 64 + c8]);
#else
        *(v8bf*)&Kt[buf ^ 1][row * 64 + c8] = *(const v8bf*)&K[g];
#endif
        vreg[r] = *(const v8bf*)&V[g];
      }
    }

    // S = scale * Q K^T : 16 rows x 64 keys per wave (8 WMMAs)
    v8f s[4];
#pragma unroll
    for (int kt = 0; kt < 4; ++kt) {
      const bf16* krow = &Kt[buf][(kt * 16) * 64];
      v16bf b0 = load_b_tile(krow, 64, lane);        // d 0..31
      v16bf b1 = load_b_tile(krow + 32, 64, lane);   // d 32..63
      v8f z;
#pragma unroll
      for (int v = 0; v < 8; ++v) z[v] = 0.f;
      z = wmma_bf16(qa0, b0, z);
      z = wmma_bf16(qa1, b1, z);
#pragma unroll
      for (int v = 0; v < 8; ++v) s[kt][v] = z[v] * scale;
    }

    // online softmax (row reductions across 16-lane halves match C layout)
#pragma unroll
    for (int v = 0; v < 8; ++v) {
      float mx = fmaxf(fmaxf(s[0][v], s[1][v]), fmaxf(s[2][v], s[3][v]));
#pragma unroll
      for (int off = 1; off < 16; off <<= 1)
        mx = fmaxf(mx, __shfl_xor(mx, off, 32));
      const float mn = fmaxf(m[v], mx);
      const float corr = __expf(m[v] - mn);
      m[v] = mn;
      float rs = 0.f;
#pragma unroll
      for (int kt = 0; kt < 4; ++kt) {
        const float e = __expf(s[kt][v] - mn);
        s[kt][v] = e;
        rs += e;
      }
#pragma unroll
      for (int off = 1; off < 16; off <<= 1)
        rs += __shfl_xor(rs, off, 32);
      l[v] = l[v] * corr + rs;
#pragma unroll
      for (int t = 0; t < 4; ++t) o[t][v] *= corr;
    }

    // C-layout -> A-layout via per-wave LDS round trip (in-order within wave)
    bf16* pw = &Pt[wave][0];
    {
      const int lh = lane >> 4;
      const int cl = lane & 15;
#pragma unroll
      for (int kt = 0; kt < 4; ++kt)
#pragma unroll
        for (int v = 0; v < 8; ++v)
          pw[(v + lh * 8) * 64 + kt * 16 + cl] = (bf16)s[kt][v];
    }
    v16bf pa0 = load_a_tile(pw, 64, lane);        // keys 0..31
    v16bf pa1 = load_a_tile(pw + 32, 64, lane);   // keys 32..63

    // O += P V : 8 WMMAs
#pragma unroll
    for (int t = 0; t < 4; ++t) {
      const bf16* vrow = &Vt[buf][(t * 16) * 64];
      v16bf b0 = load_b_tile(vrow, 64, lane);       // keys 0..31
      v16bf b1 = load_b_tile(vrow + 32, 64, lane);  // keys 32..63
      o[t] = wmma_bf16(pa0, b0, o[t]);
      o[t] = wmma_bf16(pa1, b1, o[t]);
    }

    // deferred transposed V stores for the next tile (latency hidden above)
    if (has_next) {
#pragma unroll
      for (int r = 0; r < 4; ++r) {
        const int idx = tid + r * 128;
        const int row = idx >> 3;
        const int c8 = (idx & 7) << 3;
#pragma unroll
        for (int i = 0; i < 8; ++i)
          Vt[buf ^ 1][(c8 + i) * 64 + row] = vreg[r][i];
      }
    }
  }

  const int lh = lane >> 4;
  const int cl = lane & 15;
#pragma unroll
  for (int v = 0; v < 8; ++v) {
    const float inv = 1.f / l[v];
    const size_t row = (size_t)(q0 + wave * 16 + v + lh * 8);
#pragma unroll
    for (int t = 0; t < 4; ++t)
      O[row * DMODEL + col0 + t * 16 + cl] = (bf16)(o[t][v] * inv);
  }
}

// ---------------------------------------------------------------------------
// Elementwise / LN kernels
// ---------------------------------------------------------------------------
__global__ __launch_bounds__(256) void ln_dual_kernel(
    const float* __restrict__ x,
    const float* __restrict__ gc, const float* __restrict__ bc,
    const float* __restrict__ gs, const float* __restrict__ bs,
    bf16* __restrict__ zc, bf16* __restrict__ zsb, float* __restrict__ zsf) {
  const int row = blockIdx.x * 8 + (threadIdx.x >> 5);
  const int lane = threadIdx.x & 31;
  const float* xr = x + (size_t)row * DMODEL;
  float vals[20];
  float s = 0.f;
#pragma unroll
  for (int i = 0; i < 20; ++i) { vals[i] = xr[lane + i * 32]; s += vals[i]; }
#pragma unroll
  for (int off = 1; off < 32; off <<= 1) s += __shfl_xor(s, off, 32);
  const float mu = s * (1.f / DMODEL);
  float var = 0.f;
#pragma unroll
  for (int i = 0; i < 20; ++i) { const float d = vals[i] - mu; var += d * d; }
#pragma unroll
  for (int off = 1; off < 32; off <<= 1) var += __shfl_xor(var, off, 32);
  const float rinv = rsqrtf(var * (1.f / DMODEL) + 1e-5f);
  const size_t base = (size_t)row * DMODEL;
#pragma unroll
  for (int i = 0; i < 20; ++i) {
    const int c = lane + i * 32;
    const float nx = (vals[i] - mu) * rinv;
    zc[base + c] = (bf16)(nx * gc[c] + bc[c]);
    const float xs = nx * gs[c] + bs[c];
    zsb[base + c] = (bf16)xs;
    zsf[base + c] = xs;
  }
}

__global__ void gather_kernel(const float* __restrict__ ce, bf16* __restrict__ z2, long n) {
  const long i = blockIdx.x * (long)blockDim.x + threadIdx.x;
  if (i >= n) return;
  const long row = i / DMODEL;
  const int c = (int)(i % DMODEL);
  z2[i] = (bf16)ce[row * 320 + (c >> 1)];
}

__global__ void cvt_kernel(const float* __restrict__ src, bf16* __restrict__ dst, long n) {
  const long i = blockIdx.x * (long)blockDim.x + threadIdx.x;
  if (i < n) dst[i] = (bf16)src[i];
}

__global__ void add_dual_kernel(const float* __restrict__ a, const float* __restrict__ b,
                                float* __restrict__ of, bf16* __restrict__ ob, long n) {
  const long i = blockIdx.x * (long)blockDim.x + threadIdx.x;
  if (i >= n) return;
  const float x = a[i] + b[i];
  of[i] = x;
  ob[i] = (bf16)x;
}

__global__ void final_kernel(const float* __restrict__ fus1, const float* __restrict__ up,
                             const float* __restrict__ uni1, const float* __restrict__ m2,
                             float* __restrict__ out, long n) {
  const long i = blockIdx.x * (long)blockDim.x + threadIdx.x;
  if (i < n) out[i] = 2.f * fus1[i] + up[i] + uni1[i] + m2[i];
}

// ---------------------------------------------------------------------------
// Host-side orchestration
// ---------------------------------------------------------------------------
extern "C" void kernel_launch(void* const* d_in, const int* in_sizes, int n_in,
                              void* d_out, int out_size, void* d_ws, size_t ws_size,
                              hipStream_t stream) {
  (void)in_sizes; (void)n_in; (void)out_size; (void)ws_size;
  // pytree flatten order (dicts sorted by key)
  const float* uni   = (const float*)d_in[0];
  const float* ce    = (const float*)d_in[1];
  const float* a_bd  = (const float*)d_in[2];
  const float* a_bu  = (const float*)d_in[3];
  const float* a_wd  = (const float*)d_in[4];
  const float* a_wu  = (const float*)d_in[5];
  const float* c_lnb = (const float*)d_in[6];
  const float* c_lng = (const float*)d_in[7];
  const float* c_bk  = (const float*)d_in[8];
  const float* c_bo  = (const float*)d_in[9];
  const float* c_bq  = (const float*)d_in[10];
  const float* c_bv  = (const float*)d_in[11];
  const float* c_wk  = (const float*)d_in[12];
  const float* c_wo  = (const float*)d_in[13];
  const float* c_wq  = (const float*)d_in[14];
  const float* c_wv  = (const float*)d_in[15];
  const float* m_b1  = (const float*)d_in[16];
  const float* m_b2  = (const float*)d_in[17];
  const float* m_w1  = (const float*)d_in[18];
  const float* m_w2  = (const float*)d_in[19];
  const float* s_lnb = (const float*)d_in[20];
  const float* s_lng = (const float*)d_in[21];
  const float* s_bk  = (const float*)d_in[22];
  const float* s_bo  = (const float*)d_in[23];
  const float* s_bq  = (const float*)d_in[24];
  const float* s_bv  = (const float*)d_in[25];
  const float* s_wk  = (const float*)d_in[26];
  const float* s_wo  = (const float*)d_in[27];
  const float* s_wq  = (const float*)d_in[28];
  const float* s_wv  = (const float*)d_in[29];

  char* ws = (char*)d_ws;
  size_t off = 0;
  auto alloc = [&](size_t bytes) -> void* {
    void* p = ws + off;
    off += (bytes + 255) & ~(size_t)255;
    return p;
  };
  const size_t NB = (size_t)NTOK * DMODEL;
  const size_t BF = NB * sizeof(bf16);
  const size_t FF = NB * sizeof(float);
  const size_t WSZ = (size_t)DMODEL * DMODEL;

  bf16* g_z   = (bf16*)alloc(BF);   // z (cross LN); reused: self-attn output
  bf16* g_z2  = (bf16*)alloc(BF);   // z2 gather; reused: fus1 bf16, mlp hidden
  bf16* g_xn  = (bf16*)alloc(BF);   // xn (self LN), bf16
  bf16* g_q   = (bf16*)alloc(BF);
  bf16* g_k   = (bf16*)alloc(BF);
  bf16* g_v   = (bf16*)alloc(BF);
  bf16* g_at  = (bf16*)alloc(BF);   // cross-attn output; reused: uni1 bf16
  float* xnf  = (float*)alloc(FF);  // xn f32
  float* t0   = (float*)alloc(FF);  // cattn / sattn / mlp2
  float* fus1 = (float*)alloc(FF);
  float* upf  = (float*)alloc(FF);
  float* uni1 = (float*)alloc(FF);
  bf16* down  = (bf16*)alloc((size_t)NTOK * 64 * sizeof(bf16));
  bf16* w_cq = (bf16*)alloc(WSZ * 2); bf16* w_ck = (bf16*)alloc(WSZ * 2);
  bf16* w_cv = (bf16*)alloc(WSZ * 2); bf16* w_co = (bf16*)alloc(WSZ * 2);
  bf16* w_sq = (bf16*)alloc(WSZ * 2); bf16* w_sk = (bf16*)alloc(WSZ * 2);
  bf16* w_sv = (bf16*)alloc(WSZ * 2); bf16* w_so = (bf16*)alloc(WSZ * 2);
  bf16* w_m1 = (bf16*)alloc(WSZ * 2); bf16* w_m2 = (bf16*)alloc(WSZ * 2);
  bf16* w_ad = (bf16*)alloc((size_t)64 * DMODEL * 2);
  bf16* w_au = (bf16*)alloc((size_t)DMODEL * 64 * 2);

  auto cvt = [&](const float* s, bf16* d, size_t n) {
    cvt_kernel<<<(unsigned)((n + 255) / 256), 256, 0, stream>>>(s, d, (long)n);
  };
  auto gemm640 = [&](const bf16* A, const bf16* W, const float* bias, bf16* Cb,
                     float* Cf, int nout, int relu) {
    dim3 grid(NTOK / 256, nout / 64);
    gemm_bias_kernel<640><<<grid, 256, 0, stream>>>(A, W, bias, Cb, Cf, nout, relu);
  };
  auto gemm64 = [&](const bf16* A, const bf16* W, const float* bias, bf16* Cb,
                    float* Cf, int nout, int relu) {
    dim3 grid(NTOK / 256, nout / 64);
    gemm_bias_kernel<64><<<grid, 256, 0, stream>>>(A, W, bias, Cb, Cf, nout, relu);
  };
  const unsigned EG = (unsigned)(NB / 256);

  // weights -> bf16 (L2-resident afterwards)
  cvt(c_wq, w_cq, WSZ); cvt(c_wk, w_ck, WSZ); cvt(c_wv, w_cv, WSZ); cvt(c_wo, w_co, WSZ);
  cvt(s_wq, w_sq, WSZ); cvt(s_wk, w_sk, WSZ); cvt(s_wv, w_sv, WSZ); cvt(s_wo, w_so, WSZ);
  cvt(m_w1, w_m1, WSZ); cvt(m_w2, w_m2, WSZ);
  cvt(a_wd, w_ad, (size_t)64 * DMODEL); cvt(a_wu, w_au, (size_t)DMODEL * 64);

  // LNs + gather
  ln_dual_kernel<<<NTOK / 8, 256, 0, stream>>>(uni, c_lng, c_lnb, s_lng, s_lnb,
                                               g_z, g_xn, xnf);
  gather_kernel<<<EG, 256, 0, stream>>>(ce, g_z2, (long)NB);

  // ---- cross attention ----
  gemm640(g_z2, w_cq, c_bq, g_q, nullptr, DMODEL, 0);
  gemm640(g_z,  w_ck, c_bk, g_k, nullptr, DMODEL, 0);
  gemm640(g_z,  w_cv, c_bv, g_v, nullptr, DMODEL, 0);
  flash_attn_kernel<<<dim3(NTOK / 64, NHEADS), 128, 0, stream>>>(g_q, g_k, g_v, g_at, 0.125f);
  gemm640(g_at, w_co, c_bo, nullptr, t0, DMODEL, 0);

  // fus1 = uni + cattn ;  bf16 copy feeds adapter
  add_dual_kernel<<<EG, 256, 0, stream>>>(uni, t0, fus1, g_z2, (long)NB);
  gemm640(g_z2, w_ad, a_bd, down, nullptr, 64, 1);
  gemm64(down, w_au, a_bu, nullptr, upf, DMODEL, 0);

  // ---- self attention ----
  gemm640(g_xn, w_sq, s_bq, g_q, nullptr, DMODEL, 0);
  gemm640(g_xn, w_sk, s_bk, g_k, nullptr, DMODEL, 0);
  gemm640(g_xn, w_sv, s_bv, g_v, nullptr, DMODEL, 0);
  flash_attn_kernel<<<dim3(NTOK / 64, NHEADS), 128, 0, stream>>>(g_q, g_k, g_v, g_z, 0.125f);
  gemm640(g_z, w_so, s_bo, nullptr, t0, DMODEL, 0);

  // uni1 = xn + sattn ; bf16 copy feeds MLP
  add_dual_kernel<<<EG, 256, 0, stream>>>(xnf, t0, uni1, g_at, (long)NB);
  gemm640(g_at, w_m1, m_b1, g_z2, nullptr, DMODEL, 1);
  gemm640(g_z2, w_m2, m_b2, nullptr, t0, DMODEL, 0);

  // out = 2*fus1 + up + uni1 + mlp2
  final_kernel<<<EG, 256, 0, stream>>>(fus1, upf, uni1, t0, (float*)d_out, (long)NB);
}
